// SpectralAttention_SpectralLinearAttention_44659069944189
// MI455X (gfx1250) — matrix operand order repaired and checked by reading, not compiled, not run on hardware
//
#include <hip/hip_runtime.h>
#include <cstddef>

// ---------------- problem constants ----------------
#define BB   4
#define NN   4096
#define CC   512
#define HH   8
#define DD   64
#define MM   (BB * NN)      // 16384
#define K3   (3 * CC)       // 1536
#define CSPL 16             // split-K slices for ctx GEMM
#define ACHK 32             // center chunks for gaussian stage

typedef __bf16 bf16;
typedef __attribute__((ext_vector_type(16))) __bf16 v16bf;
typedef __attribute__((ext_vector_type(8)))  __bf16 v8bf;
typedef __attribute__((ext_vector_type(8)))  float  v8f;

union AFrag { v16bf v; v8bf h[2]; bf16 e[16]; };

// A-matrix fragment: row-major source, lane holds row M = lane&15.
// elements e -> K = k0 + (lane>>4)*16 + e (contiguous)
__device__ inline v16bf load_a_frag(const bf16* __restrict__ row, int k0, int lane) {
    AFrag a;
    int sub = (lane >> 4) * 16;
    a.h[0] = *(const v8bf*)(row + k0 + sub);
    a.h[1] = *(const v8bf*)(row + k0 + 8 + sub);
    return a.v;
}

// B-matrix fragment where B[k,n] = W[n,k] (W row-major, ld = ldk).
// lane holds column n0+(lane&15); elements e -> K = k0 + (lane>>4)*16 + e (contiguous).
__device__ inline v16bf load_b_from_w_rows(const bf16* __restrict__ w, int ldk,
                                           int n0, int k0, int lane) {
    const bf16* p = w + (size_t)(n0 + (lane & 15)) * ldk + k0 + ((lane >> 4) * 16);
    return *(const v16bf*)p;
}

__device__ inline v8f wmma_bf16(v16bf a, v16bf b, v8f c) {
    return __builtin_amdgcn_wmma_f32_16x16x32_bf16(false, a, false, b, (short)0, c,
                                                   false, false);
}

// ---------------- stage 1: curvature sign counts ----------------
__global__ __launch_bounds__(256) void k_count(const float* __restrict__ x,
                                               int* __restrict__ pcnt,
                                               int* __restrict__ vcnt) {
    int i = blockIdx.x;                       // sc index, center = i+1
    __shared__ int sp, sv;
    if (threadIdx.x == 0) { sp = 0; sv = 0; }
    __syncthreads();
    int lp = 0, lv = 0;
    for (int it = threadIdx.x; it < BB * CC; it += 256) {
        int b = it >> 9, c = it & (CC - 1);
        size_t base = ((size_t)b * NN + i) * CC + c;
        float x0 = x[base], x1 = x[base + CC], x2 = x[base + 2 * CC];
        float d2 = x2 - 2.f * x1 + x0;
        if (d2 < 0.f) ++lp; else if (d2 > 0.f) ++lv;
    }
    atomicAdd(&sp, lp);
    atomicAdd(&sv, lv);
    __syncthreads();
    if (threadIdx.x == 0) { pcnt[i] = sp; vcnt[i] = sv; }
}

// ---------------- stage 2: gaussian mixture (split over center chunks) ----
__global__ __launch_bounds__(256) void k_attn(const int* __restrict__ pcnt,
                                              const int* __restrict__ vcnt,
                                              const float* __restrict__ logpw,
                                              const float* __restrict__ logvw,
                                              float* __restrict__ attnP) {
    int p  = blockIdx.x * 256 + threadIdx.x;  // position 0..4095
    int c0 = blockIdx.y * ((NN - 2 + ACHK - 1) / ACHK);
    int c1 = min(c0 + (NN - 2 + ACHK - 1) / ACHK, NN - 2);
    float ap = expf(-logpw[0]);
    float av = expf(-logvw[0]);
    float s = 0.f;
    for (int i = c0; i < c1; ++i) {
        float dc = (float)(p - (i + 1));
        float dp = dc * ap, dv = dc * av;
        s += (float)pcnt[i] * expf(-dp * dp) + (float)vcnt[i] * expf(-dv * dv);
    }
    attnP[blockIdx.y * NN + p] = s;           // deterministic partial store
}

// ---------------- stage 3: reduce partials + min-max normalize ------------
__global__ __launch_bounds__(1024) void k_norm(const float* __restrict__ attnP,
                                               float* __restrict__ gate) {
    __shared__ float smn[1024], smx[1024];
    int t = threadIdx.x;
    float v[4], mn = 3.4e38f, mx = -3.4e38f;
#pragma unroll
    for (int q = 0; q < 4; ++q) {
        int p = t + q * 1024;
        float s = 0.f;
#pragma unroll 8
        for (int sl = 0; sl < ACHK; ++sl) s += attnP[sl * NN + p];
        v[q] = s;
        mn = fminf(mn, s);
        mx = fmaxf(mx, s);
    }
    smn[t] = mn; smx[t] = mx;
    __syncthreads();
    for (int off = 512; off > 0; off >>= 1) {
        if (t < off) {
            smn[t] = fminf(smn[t], smn[t + off]);
            smx[t] = fmaxf(smx[t], smx[t + off]);
        }
        __syncthreads();
    }
    float lo = smn[0], inv = 1.f / (smx[0] - lo + 1e-6f);
#pragma unroll
    for (int q = 0; q < 4; ++q) gate[t + q * 1024] = (v[q] - lo) * inv;
}

// ---------------- stage 4: spectral PE (conv-gelu-conv) ----------------
__global__ __launch_bounds__(128) void k_pe(const float* __restrict__ omega,
                                            const float* __restrict__ w1,
                                            const float* __restrict__ b1,
                                            const float* __restrict__ w2,
                                            const float* __restrict__ b2,
                                            float* __restrict__ pe2) {
    int p = blockIdx.x;        // position 0..4095
    int t = threadIdx.x;       // 0..127 = output channel of conv1
    __shared__ float sh[2 * DD];
    int g = t >> 1;            // input channel (groups = d, 2 outs per group)
    float pv[3];
#pragma unroll
    for (int s = 0; s < 3; ++s) {
        int pp = p - 1 + s;
        float vv = 0.f;
        if (pp >= 0 && pp < NN) {
            float rel = (float)((pp >> 6) - (pp & 63));
            float om  = omega[g & 31];
            vv = (g < 32) ? sinf(rel * om) : cosf(rel * om);
        }
        pv[s] = vv;
    }
    float h = w1[t * 3 + 0] * pv[0] + w1[t * 3 + 1] * pv[1] + w1[t * 3 + 2] * pv[2] + b1[t];
    h = 0.5f * h * (1.f + erff(h * 0.70710678118654752f));   // exact gelu
    sh[t] = h;
    __syncthreads();
    if (t < DD) {
        float s = b2[t];
#pragma unroll 8
        for (int ic = 0; ic < 2 * DD; ++ic) s += w2[t * 2 * DD + ic] * sh[ic];
        pe2[t * NN + p] = s;
    }
}

__global__ void k_pegate(const float* __restrict__ pe2,
                         const float* __restrict__ wg,
                         const float* __restrict__ bg,
                         float* __restrict__ gpe) {
    __shared__ float mean[DD];
    int t = threadIdx.x;       // 0..63
    float s = 0.f;
    const float* row = pe2 + (size_t)t * NN;
#pragma unroll 8
    for (int p = 0; p < NN; ++p) s += row[p];
    mean[t] = s * (1.f / (float)NN);
    __syncthreads();
    float gacc = bg[t];
#pragma unroll 8
    for (int j = 0; j < DD; ++j) gacc += wg[t * DD + j] * mean[j];
    gpe[t] = 1.f / (1.f + expf(-gacc));
}

// ---------------- stage 5: bf16 conversions ----------------
__global__ void k_xg(const float* __restrict__ x, const float* __restrict__ gate,
                     bf16* __restrict__ xg) {
    size_t i = (size_t)blockIdx.x * blockDim.x + threadIdx.x;
    if (i >= (size_t)MM * CC) return;
    int m = (int)(i >> 9);
    xg[i] = (bf16)(x[i] * gate[m & (NN - 1)]);
}

__global__ void k_cvtw(const float* __restrict__ wq, const float* __restrict__ wp,
                       bf16* __restrict__ wqb, bf16* __restrict__ wpb) {
    int i = blockIdx.x * 256 + threadIdx.x;
    if (i < K3 * CC) wqb[i] = (bf16)wq[i];
    if (i < CC * CC) wpb[i] = (bf16)wp[i];
}

// ---------------- stage 6: QKV GEMM + fused elu/rope epilogue ----------------
// grid (K3/64, MM/128), 256 threads = 8 waves, wave tile 16x64 (4 wmma tiles)
// q and k are both written TRANSPOSED [d, N] so the ctx GEMM reads only
// contiguous 32B fragments for both operands.
__global__ __launch_bounds__(256) void k_qkv_gemm(const bf16* __restrict__ xg,
                                                  const bf16* __restrict__ wq,
                                                  const float* __restrict__ bqkv,
                                                  const float* __restrict__ pe2,
                                                  const float* __restrict__ gpe,
                                                  bf16* __restrict__ qT,
                                                  bf16* __restrict__ kT,
                                                  bf16* __restrict__ vb) {
    int lane = threadIdx.x & 31, wv = threadIdx.x >> 5;
    int m0 = blockIdx.y * 128 + wv * 16;
    int n0 = blockIdx.x * 64;
    v8f acc[4] = {};
    const bf16* arow = xg + (size_t)(m0 + (lane & 15)) * CC;
    for (int k0 = 0; k0 < CC; k0 += 32) {
        __builtin_prefetch(arow + k0 + 64, 0, 3);
        v16bf a = load_a_frag(arow, k0, lane);
#pragma unroll
        for (int j = 0; j < 4; ++j) {
            v16bf b = load_b_from_w_rows(wq, CC, n0 + j * 16, k0, lane);
            acc[j] = wmma_bf16(a, b, acc[j]);
        }
    }
    const float ropeS = 0.35355339059327373f;   // d^-0.25 = 2^-1.5
    int mrb = m0 + ((lane >> 4) * 8);
#pragma unroll
    for (int j = 0; j < 4; ++j) {
        int col  = n0 + j * 16 + (lane & 15);
        float bias = bqkv[col];
        int region = col >> 9;           // 0=q, 1=k, 2=v
        int head   = (col >> 6) & 7;
        int dch    = col & 63;
#pragma unroll
        for (int r = 0; r < 8; ++r) {
            int m = mrb + r;
            int bidx = m >> 12, n = m & (NN - 1);
            size_t bh = (size_t)bidx * HH + head;
            float val = acc[j][r] + bias;
            if (region < 2) {
                val = val > 0.f ? val + 1.f : expf(val);          // elu(x)+1
                float f = 1.f + pe2[dch * NN + n] * gpe[n >> 6] * ropeS;
                val *= f;
                if (region == 0) qT[(bh * DD + dch) * NN + n] = (bf16)val;
                else             kT[(bh * DD + dch) * NN + n] = (bf16)val;
            } else {
                vb[(bh * NN + n) * DD + dch] = (bf16)val;
            }
        }
    }
}

// ---------------- stage 7: ctx = qT k, split-K partials ----------------
// grid (32, CSPL), 128 threads = 4 waves; each block does 256 of the 4096 K.
// Both fragments are contiguous 32B loads (qT rows / kT rows).
__global__ __launch_bounds__(128) void k_ctx(const bf16* __restrict__ qT,
                                             const bf16* __restrict__ kT,
                                             float* __restrict__ P) {
    int lane = threadIdx.x & 31, wv = threadIdx.x >> 5;
    int bh = blockIdx.x, sl = blockIdx.y;
    const bf16* qbase = qT + (size_t)bh * DD * NN;
    const bf16* kbase = kT + (size_t)bh * DD * NN;
    int m0 = wv * 16;
    v8f acc[4] = {};
    const bf16* arow = qbase + (size_t)(m0 + (lane & 15)) * NN;
    int ks = sl * (NN / CSPL);
    for (int kk = ks; kk < ks + NN / CSPL; kk += 32) {
        v16bf a = load_a_frag(arow, kk, lane);
#pragma unroll
        for (int j = 0; j < 4; ++j) {
            v16bf b = load_b_from_w_rows(kbase, NN, j * 16, kk, lane);
            acc[j] = wmma_bf16(a, b, acc[j]);
        }
    }
    float* pb = P + ((size_t)(sl * 32 + bh) << 12);
    int mrb = m0 + ((lane >> 4) * 8);
#pragma unroll
    for (int j = 0; j < 4; ++j) {
        int col = j * 16 + (lane & 15);
#pragma unroll
        for (int r = 0; r < 8; ++r)
            pb[(mrb + r) * DD + col] = acc[j][r];
    }
}

// reduce split-K partials, apply scale, store ctx TRANSPOSED [e, dd] so the
// out-GEMM B fragments are contiguous.
__global__ void k_ctxred(const float* __restrict__ P, bf16* __restrict__ ctxT) {
    int idx = blockIdx.x * 256 + threadIdx.x;   // 0 .. 32*4096-1
    int bh = idx >> 12, rem = idx & 4095;
    int dd = rem >> 6, e = rem & 63;
    float s = 0.f;
#pragma unroll
    for (int sl = 0; sl < CSPL; ++sl) s += P[((size_t)(sl * 32 + bh) << 12) + rem];
    ctxT[((size_t)bh << 12) + e * DD + dd] = (bf16)(s * (1.f / 512.f));
}

// ---------------- stage 8: out = v ctx * scale, merge heads ----------------
// grid (32, 64), 128 threads = 4 waves, wave tile 16x64, K = 64
__global__ __launch_bounds__(128) void k_out(const bf16* __restrict__ vb,
                                             const bf16* __restrict__ ctxT,
                                             bf16* __restrict__ om) {
    int lane = threadIdx.x & 31, wv = threadIdx.x >> 5;
    int bh = blockIdx.x;
    int b = bh >> 3, hh = bh & 7;
    const bf16* vbase = vb + (size_t)bh * NN * DD;
    const bf16* cbase = ctxT + ((size_t)bh << 12);
    int m0 = blockIdx.y * 64 + wv * 16;
    v8f acc[4] = {};
    const bf16* arow = vbase + (size_t)(m0 + (lane & 15)) * DD;
#pragma unroll
    for (int kk = 0; kk < DD; kk += 32) {
        v16bf a = load_a_frag(arow, kk, lane);
#pragma unroll
        for (int j = 0; j < 4; ++j) {
            v16bf bfr = load_b_from_w_rows(cbase, DD, j * 16, kk, lane);
            acc[j] = wmma_bf16(a, bfr, acc[j]);
        }
    }
    const float sc = 1.f / 512.f;
    int mrb = m0 + ((lane >> 4) * 8);
#pragma unroll
    for (int j = 0; j < 4; ++j) {
        int col = j * 16 + (lane & 15);
#pragma unroll
        for (int r = 0; r < 8; ++r) {
            int n = mrb + r;
            om[((size_t)b * NN + n) * CC + hh * DD + col] = (bf16)(acc[j][r] * sc);
        }
    }
}

// ---------------- stage 9: output projection ----------------
// grid (CC/64, MM/128), 256 threads = 8 waves
__global__ __launch_bounds__(256) void k_proj(const bf16* __restrict__ om,
                                              const bf16* __restrict__ wp,
                                              const float* __restrict__ bp,
                                              float* __restrict__ out) {
    int lane = threadIdx.x & 31, wv = threadIdx.x >> 5;
    int m0 = blockIdx.y * 128 + wv * 16;
    int n0 = blockIdx.x * 64;
    v8f acc[4] = {};
    const bf16* arow = om + (size_t)(m0 + (lane & 15)) * CC;
    for (int k0 = 0; k0 < CC; k0 += 32) {
        __builtin_prefetch(arow + k0 + 64, 0, 3);
        v16bf a = load_a_frag(arow, k0, lane);
#pragma unroll
        for (int j = 0; j < 4; ++j) {
            v16bf b = load_b_from_w_rows(wp, CC, n0 + j * 16, k0, lane);
            acc[j] = wmma_bf16(a, b, acc[j]);
        }
    }
    int mrb = m0 + ((lane >> 4) * 8);
#pragma unroll
    for (int j = 0; j < 4; ++j) {
        int col = n0 + j * 16 + (lane & 15);
        float bias = bp[col];
#pragma unroll
        for (int r = 0; r < 8; ++r)
            out[(size_t)(mrb + r) * CC + col] = acc[j][r] + bias;
    }
}

// ---------------- host launcher ----------------
extern "C" void kernel_launch(void* const* d_in, const int* in_sizes, int n_in,
                              void* d_out, int out_size, void* d_ws, size_t ws_size,
                              hipStream_t stream) {
    const float* x     = (const float*)d_in[0];
    const float* logpw = (const float*)d_in[1];
    const float* logvw = (const float*)d_in[2];
    const float* wqkv  = (const float*)d_in[3];
    const float* bqkv  = (const float*)d_in[4];
    const float* wproj = (const float*)d_in[5];
    const float* bproj = (const float*)d_in[6];
    const float* omega = (const float*)d_in[7];
    const float* w_pp1 = (const float*)d_in[8];
    const float* b_pp1 = (const float*)d_in[9];
    const float* w_pp2 = (const float*)d_in[10];
    const float* b_pp2 = (const float*)d_in[11];
    const float* w_gate = (const float*)d_in[12];
    const float* b_gate = (const float*)d_in[13];

    char* w = (char*)d_ws;
    size_t off = 0;
    auto carve = [&](size_t bytes) -> void* {
        void* p = w + off;
        off = (off + bytes + 255) & ~(size_t)255;
        return p;
    };
    int*   pcnt   = (int*)  carve((NN - 2) * sizeof(int));
    int*   vcnt   = (int*)  carve((NN - 2) * sizeof(int));
    float* attnP  = (float*)carve((size_t)ACHK * NN * sizeof(float));
    float* gateA  = (float*)carve(NN * sizeof(float));
    float* pe2    = (float*)carve((size_t)DD * NN * sizeof(float));
    float* gpe    = (float*)carve(DD * sizeof(float));
    bf16*  xg     = (bf16*) carve((size_t)MM * CC * sizeof(bf16));
    bf16*  wqb    = (bf16*) carve((size_t)K3 * CC * sizeof(bf16));
    bf16*  wpb    = (bf16*) carve((size_t)CC * CC * sizeof(bf16));
    bf16*  qT     = (bf16*) carve((size_t)BB * HH * DD * NN * sizeof(bf16));
    bf16*  kT     = (bf16*) carve((size_t)BB * HH * DD * NN * sizeof(bf16));
    bf16*  vb     = (bf16*) carve((size_t)BB * HH * NN * DD * sizeof(bf16));
    float* ctxP   = (float*)carve((size_t)CSPL * BB * HH * DD * DD * sizeof(float));
    bf16*  ctxT   = (bf16*) carve((size_t)BB * HH * DD * DD * sizeof(bf16));
    bf16*  om     = (bf16*) carve((size_t)MM * CC * sizeof(bf16));
    (void)ws_size; (void)n_in; (void)in_sizes; (void)out_size;

    k_count<<<NN - 2, 256, 0, stream>>>(x, pcnt, vcnt);
    k_attn<<<dim3(NN / 256, ACHK), 256, 0, stream>>>(pcnt, vcnt, logpw, logvw, attnP);
    k_norm<<<1, 1024, 0, stream>>>(attnP, gateA);
    k_pe<<<NN, 128, 0, stream>>>(omega, w_pp1, b_pp1, w_pp2, b_pp2, pe2);
    k_pegate<<<1, DD, 0, stream>>>(pe2, w_gate, b_gate, gpe);
    k_xg<<<(MM * CC) / 256, 256, 0, stream>>>(x, gateA, xg);
    k_cvtw<<<(K3 * CC) / 256, 256, 0, stream>>>(wqkv, wproj, wqb, wpb);
    k_qkv_gemm<<<dim3(K3 / 64, MM / 128), 256, 0, stream>>>(xg, wqb, bqkv, pe2, gpe,
                                                            qT, kT, vb);
    k_ctx<<<dim3(BB * HH, CSPL), 128, 0, stream>>>(qT, kT, ctxP);
    k_ctxred<<<(BB * HH * DD * DD) / 256, 256, 0, stream>>>(ctxP, ctxT);
    k_out<<<dim3(BB * HH, NN / 64), 128, 0, stream>>>(vb, ctxT, om);
    k_proj<<<dim3(CC / 64, MM / 128), 256, 0, stream>>>(om, wpb, bproj, (float*)d_out);
}
